// MaskedSelfAttentionHead_88923002896432
// MI455X (gfx1250) — matrix operand order repaired and checked by reading, not compiled, not run on hardware
//
#include <hip/hip_runtime.h>
#include <stdint.h>

#define B_  8
#define T_  2048
#define C_  1024
#define H_  128
#define BT_ (B_ * T_)

typedef __bf16 v16bf __attribute__((ext_vector_type(16)));
typedef __bf16 v8bf  __attribute__((ext_vector_type(8)));
typedef float  v8f   __attribute__((ext_vector_type(8)));

__device__ __forceinline__ __bf16 f2bf(float f) { return (__bf16)f; }

__device__ __forceinline__ v16bf cat8(v8bf lo, v8bf hi) {
    return __builtin_shufflevector(lo, hi, 0,1,2,3,4,5,6,7,8,9,10,11,12,13,14,15);
}

// ---------------------------------------------------------------------------
// Kernel 0: coalesced x f32 -> bf16 (float4 reads, b128 writes).
// ---------------------------------------------------------------------------
__global__ __launch_bounds__(256)
void x_to_bf16(const float* __restrict__ x, __bf16* __restrict__ xb) {
    size_t i = ((size_t)blockIdx.x * 256 + threadIdx.x) * 8;
    float4 a = *(const float4*)(x + i);
    float4 b = *(const float4*)(x + i + 4);
    v8bf o;
    o[0] = f2bf(a.x); o[1] = f2bf(a.y); o[2] = f2bf(a.z); o[3] = f2bf(a.w);
    o[4] = f2bf(b.x); o[5] = f2bf(b.y); o[6] = f2bf(b.z); o[7] = f2bf(b.w);
    *(v8bf*)(xb + i) = o;
}

// ---------------------------------------------------------------------------
// Kernel 1: W[C,H] f32 -> Wt[H,C] bf16 for all three weights (tiny).
// ---------------------------------------------------------------------------
__global__ __launch_bounds__(256)
void w_transpose(const float* __restrict__ Wq, const float* __restrict__ Wk,
                 const float* __restrict__ Wv, __bf16* __restrict__ wtq,
                 __bf16* __restrict__ wtk, __bf16* __restrict__ wtv) {
    int idx = blockIdx.x * 256 + threadIdx.x;        // over 3*H*C
    int m   = idx / (H_ * C_);
    int r   = idx % (H_ * C_);
    int h   = r / C_;
    int c   = r % C_;
    const float* W = (m == 0) ? Wq : (m == 1) ? Wk : Wv;
    __bf16*     Wt = (m == 0) ? wtq : (m == 1) ? wtk : wtv;
    Wt[h * C_ + c] = f2bf(W[c * H_ + h]);            // write coalesced over c
}

// ---------------------------------------------------------------------------
// Kernel 2: QKV projection with cross-iteration double buffering: operands
// for K-chunk kc+32 are loaded before the WMMAs of chunk kc execute, so the
// scheduler cannot collapse loads onto their uses and steady-state waits
// become partial.  q is pre-scaled by H^-0.5; V stored transposed [H, BT].
// ---------------------------------------------------------------------------
__global__ __launch_bounds__(96)
void qkv_proj_wmma(const __bf16* __restrict__ xb,
                   const __bf16* __restrict__ wtq,
                   const __bf16* __restrict__ wtk,
                   const __bf16* __restrict__ wtv,
                   __bf16* __restrict__ qb,
                   __bf16* __restrict__ kb,
                   __bf16* __restrict__ vbt)
{
    const int lane = threadIdx.x & 31;
    const int wv   = threadIdx.x >> 5;               // 0=q 1=k 2=v
    const int hi   = lane >> 4;
    const int l16  = lane & 15;
    const int rt   = blockIdx.x;

    const __bf16* Wt = (wv == 0) ? wtq : (wv == 1) ? wtk : wtv;
    const float oscale = (wv == 0) ? 0.08838834764831845f : 1.0f;

    v8f acc[8] = {};
    const __bf16* xrow = xb + (size_t)(rt * 16 + l16) * C_;
    const __bf16* wrow = Wt + (size_t)l16 * C_;      // + n*16*C_

    auto loadA = [&](v16bf& aa, int kc) {
        v8bf lo = *(const v8bf*)(xrow + kc + hi * 16);
        v8bf h8 = *(const v8bf*)(xrow + kc + hi * 16 + 8);
        aa = cat8(lo, h8);
    };
    auto loadB = [&](v16bf* bb, int kc) {
#pragma unroll
        for (int n = 0; n < 8; ++n)
            bb[n] = *(const v16bf*)(wrow + (size_t)n * 16 * C_ + kc + hi * 16);
    };
    auto domma = [&](v16bf& aa, v16bf* bb) {
#pragma unroll
        for (int n = 0; n < 8; ++n)
            acc[n] = __builtin_amdgcn_wmma_f32_16x16x32_bf16(
                false, aa, false, bb[n], (short)0, acc[n], false, false);
    };

    v16bf a0, a1, b0[8], b1[8];
    loadA(a0, 0);
    loadB(b0, 0);
    for (int kc = 0; kc < C_; kc += 64) {
        loadA(a1, kc + 32);                  // prefetch next half-step
        loadB(b1, kc + 32);
        domma(a0, b0);                       // consume buffers loaded earlier
        if (kc + 64 < C_) {                  // prefetch step kc+64
            loadA(a0, kc + 64);
            loadB(b0, kc + 64);
        }
        domma(a1, b1);
    }

    if (wv == 2) {                                    // V: store transposed
#pragma unroll
        for (int n = 0; n < 8; ++n)
#pragma unroll
            for (int r = 0; r < 8; ++r)
                vbt[(size_t)(n * 16 + l16) * BT_ + rt * 16 + r + hi * 8] =
                    f2bf(acc[n][r]);
    } else {
        __bf16* out = (wv == 0) ? qb : kb;
#pragma unroll
        for (int n = 0; n < 8; ++n)
#pragma unroll
            for (int r = 0; r < 8; ++r)
                out[(size_t)(rt * 16 + r + hi * 8) * H_ + n * 16 + l16] =
                    f2bf(acc[n][r] * oscale);
    }
}

// ---------------------------------------------------------------------------
// Kernel 3: flash attention, one wave per 16-row query tile, 4 waves/block.
// K+V tiles for step st+1 are loaded before the S-WMMA / softmax / PV chain
// of step st (ping-pong buffers), overlapping global latency with compute.
// ---------------------------------------------------------------------------
__global__ __launch_bounds__(128)
void attn_flash_wmma(const __bf16* __restrict__ qb,
                     const __bf16* __restrict__ kb,
                     const __bf16* __restrict__ vbt,
                     float* __restrict__ out)
{
    __shared__ alignas(32) __bf16 plds[4][16 * 32];
    const int lane = threadIdx.x & 31;
    const int wv   = threadIdx.x >> 5;
    const int hi   = lane >> 4;
    const int l16  = lane & 15;
    const int qt   = blockIdx.x * 4 + wv;
    const int b    = qt >> 7;
    const int tt   = qt & 127;
    const int t0   = tt * 16;
    const long base = (long)b * T_;
    __bf16* pl = plds[wv];

    // Q in A-layout: per chunk, two contiguous 8-element runs
    v16bf qa[4];
    const __bf16* qrow = qb + (size_t)(base + t0 + l16) * H_;
#pragma unroll
    for (int c = 0; c < 4; ++c) {
        v8bf qlo = *(const v8bf*)(qrow + c * 32 + hi * 16);
        v8bf qhi = *(const v8bf*)(qrow + c * 32 + hi * 16 + 8);
        qa[c] = cat8(qlo, qhi);
    }

    v8f o[8] = {};
    float mr[8], lr[8];
#pragma unroll
    for (int r = 0; r < 8; ++r) { mr[r] = -3.0e38f; lr[r] = 0.0f; }

    const __bf16* vcol = vbt + (size_t)base + (size_t)l16 * BT_;  // + n*16*BT_

    auto loadKV = [&](v16bf* kt, v16bf* vt, int s0) {
        const __bf16* krow0 = kb + (size_t)(base + s0 + l16) * H_;
        const __bf16* krow1 = krow0 + 16 * H_;
#pragma unroll
        for (int c = 0; c < 4; ++c) {
            kt[c]     = *(const v16bf*)(krow0 + c * 32 + hi * 16);
            kt[c + 4] = *(const v16bf*)(krow1 + c * 32 + hi * 16);
        }
#pragma unroll
        for (int n = 0; n < 8; ++n)
            vt[n] = *(const v16bf*)(vcol + (size_t)n * 16 * BT_ + s0 + hi * 16);
    };

    auto process = [&](int s0, v16bf* kt, v16bf* vt) {
        // ---- S = Q K^T ----
        v8f slo = {}, shi = {};
#pragma unroll
        for (int c = 0; c < 4; ++c) {
            slo = __builtin_amdgcn_wmma_f32_16x16x32_bf16(
                false, qa[c], false, kt[c], (short)0, slo, false, false);
            shi = __builtin_amdgcn_wmma_f32_16x16x32_bf16(
                false, qa[c], false, kt[c + 4], (short)0, shi, false, false);
        }
        // causal mask
#pragma unroll
        for (int r = 0; r < 8; ++r) {
            int t = t0 + r + hi * 8;
            if (s0 + l16      > t) slo[r] = -3.0e38f;
            if (s0 + 16 + l16 > t) shi[r] = -3.0e38f;
        }
        // online softmax over the 32 new columns
        float alpha[8];
        v8f plo, phi;
#pragma unroll
        for (int r = 0; r < 8; ++r) {
            float v = fmaxf(slo[r], shi[r]);
            v = fmaxf(v, __shfl_xor(v, 1, 32));
            v = fmaxf(v, __shfl_xor(v, 2, 32));
            v = fmaxf(v, __shfl_xor(v, 4, 32));
            v = fmaxf(v, __shfl_xor(v, 8, 32));
            float mn = fmaxf(mr[r], v);
            alpha[r] = __expf(mr[r] - mn);
            float p0 = __expf(slo[r] - mn);
            float p1 = __expf(shi[r] - mn);
            plo[r] = p0; phi[r] = p1;
            float s = p0 + p1;
            s += __shfl_xor(s, 1, 32);
            s += __shfl_xor(s, 2, 32);
            s += __shfl_xor(s, 4, 32);
            s += __shfl_xor(s, 8, 32);
            lr[r] = lr[r] * alpha[r] + s;
            mr[r] = mn;
        }
#pragma unroll
        for (int n = 0; n < 8; ++n)
#pragma unroll
            for (int r = 0; r < 8; ++r) o[n][r] *= alpha[r];
        // stage P (C-layout) -> LDS row-major 16x32 bf16
#pragma unroll
        for (int r = 0; r < 8; ++r) {
            int row = r + hi * 8;
            pl[row * 32 + l16]      = f2bf(plo[r]);
            pl[row * 32 + 16 + l16] = f2bf(phi[r]);
        }
        __builtin_amdgcn_fence(__ATOMIC_ACQ_REL, "wavefront");
        // reload P in A-layout: two contiguous 8-element DS loads
        v8bf pl8 = *(const v8bf*)(pl + l16 * 32 + hi * 16);
        v8bf ph8 = *(const v8bf*)(pl + l16 * 32 + hi * 16 + 8);
        v16bf pa = cat8(pl8, ph8);
        // ---- O += P @ V ----
#pragma unroll
        for (int n = 0; n < 8; ++n)
            o[n] = __builtin_amdgcn_wmma_f32_16x16x32_bf16(
                false, pa, false, vt[n], (short)0, o[n], false, false);
    };

    const int nsteps = (tt >> 1) + 1;
    v16bf ktA[8], vtA[8], ktB[8], vtB[8];
    loadKV(ktA, vtA, 0);
    for (int st = 0; st < nsteps; st += 2) {
        if (st + 1 < nsteps) loadKV(ktB, vtB, (st + 1) * 32);
        process(st * 32, ktA, vtA);
        if (st + 1 >= nsteps) break;
        if (st + 2 < nsteps) loadKV(ktA, vtA, (st + 2) * 32);
        process((st + 1) * 32, ktB, vtB);
    }

    // normalize (v_rcp + mul instead of 64 full divisions) and write fp32
    float inv[8];
#pragma unroll
    for (int r = 0; r < 8; ++r) inv[r] = __builtin_amdgcn_rcpf(lr[r]);
#pragma unroll
    for (int n = 0; n < 8; ++n)
#pragma unroll
        for (int r = 0; r < 8; ++r) {
            int t = t0 + r + hi * 8;
            out[(base + t) * H_ + n * 16 + l16] = o[n][r] * inv[r];
        }
}

extern "C" void kernel_launch(void* const* d_in, const int* in_sizes, int n_in,
                              void* d_out, int out_size, void* d_ws, size_t ws_size,
                              hipStream_t stream) {
    const float* x  = (const float*)d_in[0];
    const float* Wq = (const float*)d_in[1];
    const float* Wk = (const float*)d_in[2];
    const float* Wv = (const float*)d_in[3];
    float* out = (float*)d_out;

    const size_t nx   = (size_t)BT_ * C_;            // 16.7M
    const size_t nqkv = (size_t)BT_ * H_;            // 2M
    const size_t nw   = (size_t)C_ * H_;             // 128K
    __bf16* xb  = (__bf16*)d_ws;
    __bf16* qb  = xb + nx;
    __bf16* kb  = qb + nqkv;
    __bf16* vbt = kb + nqkv;
    __bf16* wtq = vbt + nqkv;
    __bf16* wtk = wtq + nw;
    __bf16* wtv = wtk + nw;

    x_to_bf16 <<<nx / (256 * 8), 256, 0, stream>>>(x, xb);
    w_transpose<<<(3 * nw) / 256, 256, 0, stream>>>(Wq, Wk, Wv, wtq, wtk, wtv);
    qkv_proj_wmma<<<BT_ / 16, 96, 0, stream>>>(xb, wtq, wtk, wtv, qb, kb, vbt);
    attn_flash_wmma<<<BT_ / 16 / 4, 128, 0, stream>>>(qb, kb, vbt, out);
}
